// MFWEncdecMultiheadAttn_26989574488823
// MI455X (gfx1250) — matrix operand order repaired and checked by reading, not compiled, not run on hardware
//
#include <hip/hip_runtime.h>
#include <hip/hip_bf16.h>

// ---------------- vector types for WMMA fragments ----------------
typedef __attribute__((ext_vector_type(16))) _Float16 v16h;
typedef __attribute__((ext_vector_type(8)))  float    v8f;
typedef __attribute__((ext_vector_type(8)))  _Float16 h8v;   // 16 bytes

// Problem constants
#define E_DIM   1024
#define H_HEADS 16
#define D_HEAD  64
#define SEQ     1024
#define BATCH   8
#define NLANG   8
#define RANKLR  4
#define OUT_ELEMS (SEQ * BATCH * E_DIM)   // 8388608 floats in d_out before probs

// =================================================================
// Fragment loaders per CDNA5 ISA §7.12.2 (wave32 layouts)
// =================================================================

// A-matrix 16x32 f16 (MxK): lane<16 row=lane holds K {0..7} in v0..3, {16..23} in v4..7
//                           lane>=16 same row holds K {8..15}, {24..31}
__device__ inline v16h load_fragA(const _Float16* __restrict__ p, long ld, int lane) {
  int r = lane & 15;
  int k = (lane < 16) ? 0 : 8;
  const _Float16* q = p + (long)r * ld + k;
  union { v16h v; h8v h[2]; } u;
  u.h[0] = *(const h8v*)(q);
  u.h[1] = *(const h8v*)(q + 16);
  return u.v;
}

// Same A layout, but source is fp32 (probs) -> convert in-register
__device__ inline v16h load_fragA(const float* __restrict__ p, long ld, int lane) {
  int r = lane & 15;
  int k = (lane < 16) ? 0 : 8;
  const float* q = p + (long)r * ld + k;
  union { v16h v; _Float16 e[16]; } u;
#pragma unroll
  for (int i = 0; i < 8; ++i) u.e[i]     = (_Float16)q[i];
#pragma unroll
  for (int i = 0; i < 8; ++i) u.e[8 + i] = (_Float16)q[16 + i];
  return u.v;
}

// B-matrix 32x16 f16 (KxN), B = W^T so column n of B == row n of W (row-major [N][K]):
// lane<16 col=lane holds K {0..7} in v0..3, {16..23} in v4..7; lane>=16 holds K {8..15}, {24..31}
__device__ inline v16h load_fragB(const _Float16* __restrict__ p, long ld, int lane) {
  int n = lane & 15;
  int k = (lane < 16) ? 0 : 8;
  const _Float16* q = p + (long)n * ld + k;
  union { v16h v; h8v h[2]; } u;
  u.h[0] = *(const h8v*)(q);
  u.h[1] = *(const h8v*)(q + 16);
  return u.v;
}

// C/D 16x16 f32: VGPR j -> row j (lanes 0-15) / row j+8 (lanes 16-31), col = lane%16
__device__ inline void store_frag(float* __restrict__ C, long ldc, int row0, int col0,
                                  int lane, v8f c) {
  int col = col0 + (lane & 15);
  int r8  = row0 + ((lane >> 4) << 3);
#pragma unroll
  for (int j = 0; j < 8; ++j) C[(long)(r8 + j) * ldc + col] = c[j];
}

__device__ inline void store_frag(_Float16* __restrict__ C, long ldc, int row0, int col0,
                                  int lane, v8f c) {
  int col = col0 + (lane & 15);
  int r8  = row0 + ((lane >> 4) << 3);
#pragma unroll
  for (int j = 0; j < 8; ++j) C[(long)(r8 + j) * ldc + col] = (_Float16)c[j];
}

// =================================================================
// Generic GEMM: C[m,n] = sum_k A[m,k] * B[n,k]
// A row-major [M][K] f16, B row-major [N][K] f16, fp32 accum via
// v_wmma_f32_16x16x32_f16, C f16 or f32. Wave tile 64x32 (4x2 frags).
// =================================================================
template<int BM, int BN, typename CT>
__global__ void __launch_bounds__((BM / 64) * (BN / 32) * 32)
gemm_bt(const _Float16* __restrict__ A, long lda,
        const _Float16* __restrict__ B, long ldb,
        CT* __restrict__ C, long ldc, int K) {
  constexpr int WAVES_N = BN / 32;
  const int wave = threadIdx.x >> 5;
  const int lane = threadIdx.x & 31;
  const int wm = wave / WAVES_N;
  const int wn = wave % WAVES_N;

  const int row0 = blockIdx.y * BM + wm * 64;
  const int col0 = blockIdx.x * BN + wn * 32;

  const _Float16* Ab = A + (long)row0 * lda;
  const _Float16* Bb = B + (long)col0 * ldb;

  v8f acc[4][2] = {};

  for (int k0 = 0; k0 < K; k0 += 32) {
    v16h af[4];
#pragma unroll
    for (int i = 0; i < 4; ++i)
      af[i] = load_fragA(Ab + (long)(i * 16) * lda + k0, lda, lane);
    v16h bf[2];
#pragma unroll
    for (int j = 0; j < 2; ++j)
      bf[j] = load_fragB(Bb + (long)(j * 16) * ldb + k0, ldb, lane);
#pragma unroll
    for (int i = 0; i < 4; ++i)
#pragma unroll
      for (int j = 0; j < 2; ++j)
        acc[i][j] = __builtin_amdgcn_wmma_f32_16x16x32_f16(
            false, af[i], false, bf[j], (short)0, acc[i][j], false, false);
  }

#pragma unroll
  for (int i = 0; i < 4; ++i)
#pragma unroll
    for (int j = 0; j < 2; ++j)
      store_frag(C, ldc, row0 + i * 16, col0 + j * 16, lane, acc[i][j]);
}

// =================================================================
// Fused attention: per block = (b*16+h, 128 q-rows)
//   A: raw scores S = q_tile @ k^T  -> probs region (fp32, L2-hot)
//   B: softmax in-register per wave (rows stay inside this block)
//   C: ctx = P @ v^T, P re-read fp32->f16 from hot L2
// 256 threads = 8 waves. No divergence around WMMA (EXEC all ones).
// =================================================================
__global__ void __launch_bounds__(256)
fused_attn(const _Float16* __restrict__ qf, const _Float16* __restrict__ kvf,
           const _Float16* __restrict__ vt, float* __restrict__ probs,
           _Float16* __restrict__ ctx) {
  const int z = blockIdx.z;           // b*16 + h
  const int b = z >> 4, h = z & 15;
  const int qt = blockIdx.y;          // which 128-row q tile
  const int wave = threadIdx.x >> 5;
  const int lane = threadIdx.x & 31;

  const long aBase = (long)b * 1024 + (long)h * 64;   // into qf,  lda = 8192
  const long kBase = (long)b * 2048 + (long)h * 64;   // into kvf, ldb = 16384
  float* P = probs + (long)z * 1048576;               // [1024][1024]
  const int qrow0 = qt * 128;

  // ---- Phase A: raw scores (128 x 1024), K = 64 ----
  {
    const int wm = wave >> 2;         // 0..1 -> 64-row half
    const int wn = wave & 3;          // 0..3 -> 32-col slot
    const int row0 = qrow0 + wm * 64;

    v16h af[2][4];                    // cache q frags for both k-steps
#pragma unroll
    for (int ks = 0; ks < 2; ++ks)
#pragma unroll
      for (int i = 0; i < 4; ++i)
        af[ks][i] = load_fragA(qf + aBase + (long)(row0 + i * 16) * 8192 + ks * 32,
                               8192, lane);

    for (int nt = 0; nt < 8; ++nt) {  // 8 x 128 key columns
      const int col0 = nt * 128 + wn * 32;
      v8f acc[4][2] = {};
#pragma unroll
      for (int ks = 0; ks < 2; ++ks) {
        v16h bf[2];
#pragma unroll
        for (int j = 0; j < 2; ++j)
          bf[j] = load_fragB(kvf + kBase + (long)(col0 + j * 16) * 16384 + ks * 32,
                             16384, lane);
#pragma unroll
        for (int i = 0; i < 4; ++i)
#pragma unroll
          for (int j = 0; j < 2; ++j)
            acc[i][j] = __builtin_amdgcn_wmma_f32_16x16x32_f16(
                false, af[ks][i], false, bf[j], (short)0, acc[i][j], false, false);
      }
#pragma unroll
      for (int i = 0; i < 4; ++i)
#pragma unroll
        for (int j = 0; j < 2; ++j)
          store_frag(P, 1024, row0 + i * 16, col0 + j * 16, lane, acc[i][j]);
    }
  }
  __threadfence();     // publish raw scores + invalidate L0 for re-read
  __syncthreads();

  // ---- Phase B: softmax. wave -> 16 rows; 32 lanes x 32 cols per row ----
  {
    for (int rr = 0; rr < 16; ++rr) {
      const int row = qrow0 + wave * 16 + rr;
      float* pr = P + (long)row * 1024;
      float v[32];
      float m = -__FLT_MAX__;
#pragma unroll
      for (int i = 0; i < 32; ++i) { v[i] = pr[lane + 32 * i]; m = fmaxf(m, v[i]); }
#pragma unroll
      for (int s = 16; s > 0; s >>= 1) m = fmaxf(m, __shfl_xor(m, s, 32));
      float sum = 0.f;
#pragma unroll
      for (int i = 0; i < 32; ++i) { v[i] = __expf(v[i] - m); sum += v[i]; }
#pragma unroll
      for (int s = 16; s > 0; s >>= 1) sum += __shfl_xor(sum, s, 32);
      const float inv = 1.0f / sum;
#pragma unroll
      for (int i = 0; i < 32; ++i) pr[lane + 32 * i] = v[i] * inv;
    }
  }
  __threadfence();     // publish normalized probs + drop stale raw-score L0 lines
  __syncthreads();

  // ---- Phase C: ctx tile (128 x 64) = P (fp32->f16) @ vt^T, wave tile 64x16 ----
  {
    const int wm = wave >> 2;         // 0..1 -> 64-row half
    const int wn = wave & 3;          // 0..3 -> 16-col (d) slot
    const int row0 = qrow0 + wm * 64;
    const int col0 = wn * 16;
    const _Float16* V = vt + (long)z * 65536;   // [64][1024] per head

    v8f acc[4] = {};
    for (int k0 = 0; k0 < 1024; k0 += 32) {
      v16h afr[4];
#pragma unroll
      for (int i = 0; i < 4; ++i)
        afr[i] = load_fragA(P + (long)(row0 + i * 16) * 1024 + k0, 1024, lane);
      v16h bfr = load_fragB(V + (long)col0 * 1024 + k0, 1024, lane);
#pragma unroll
      for (int i = 0; i < 4; ++i)
        acc[i] = __builtin_amdgcn_wmma_f32_16x16x32_f16(
            false, afr[i], false, bfr, (short)0, acc[i], false, false);
    }
    _Float16* Cb = ctx + (long)b * 1024 + (long)h * 64;   // ldc = 8192
#pragma unroll
    for (int i = 0; i < 4; ++i)
      store_frag(Cb, 8192, row0 + i * 16, col0, lane, acc[i]);
  }
}

// =================================================================
// W = base * (rm^T sm) + (r^T s), optionally pre-scaled, cast to f16
// =================================================================
__global__ void build_w(const float* __restrict__ base,
                        const float* __restrict__ r,  const float* __restrict__ s,
                        const float* __restrict__ rm, const float* __restrict__ sm,
                        const int* __restrict__ tgt, const int* __restrict__ src,
                        _Float16* __restrict__ W, int OUT, int IN, float scale) {
  long idx = (long)blockIdx.x * blockDim.x + threadIdx.x;
  long total = (long)OUT * IN;
  if (idx >= total) return;
  int o = (int)(idx / IN);
  int i = (int)(idx % IN);
  int it = tgt[0], is = src[0];
  float lr = 0.f, lrm = 0.f;
#pragma unroll
  for (int rr = 0; rr < RANKLR; ++rr) {
    lrm += rm[((long)it * RANKLR + rr) * OUT + o] * sm[((long)is * RANKLR + rr) * IN + i];
    lr  += r [((long)it * RANKLR + rr) * OUT + o] * s [((long)is * RANKLR + rr) * IN + i];
  }
  float w = base[idx] * lrm + lr;
  W[idx] = (_Float16)(w * scale);
}

__global__ void cast_f16(const float* __restrict__ x, _Float16* __restrict__ y, long n) {
  long idx = (long)blockIdx.x * blockDim.x + threadIdx.x;
  if (idx < n) y[idx] = (_Float16)x[idx];
}

// =================================================================
// vt[(b*16+h)*64 + d][l] = kv[(l*8+b)*2048 + 1024 + h*64 + d]  (LDS-tiled)
// =================================================================
__global__ void transpose_v(const _Float16* __restrict__ kv, _Float16* __restrict__ vt) {
  __shared__ _Float16 tile[64][65];
  const int bh = blockIdx.y;
  const int lt = blockIdx.x;
  const int b = bh >> 4, h = bh & 15;
  const int t = threadIdx.x;
  for (int idx = t; idx < 4096; idx += 256) {
    int r = idx >> 6, c = idx & 63;                   // r = local l, c = d
    long l = (long)lt * 64 + r;
    tile[c][r] = kv[(l * 8 + b) * 2048 + 1024 + h * 64 + c];
  }
  __syncthreads();
  for (int idx = t; idx < 4096; idx += 256) {
    int r = idx >> 6, c = idx & 63;                   // r = d, c = local l
    vt[((long)bh * 64 + r) * 1024 + (long)lt * 64 + c] = tile[r][c];
  }
}

// =================================================================
// Host-side orchestration
// =================================================================
extern "C" void kernel_launch(void* const* d_in, const int* in_sizes, int n_in,
                              void* d_out, int out_size, void* d_ws, size_t ws_size,
                              hipStream_t stream) {
  const float* query = (const float*)d_in[0];
  const float* key   = (const float*)d_in[1];
  const float* Wq0   = (const float*)d_in[2];
  const float* Wkv0  = (const float*)d_in[3];
  const float* Wo0   = (const float*)d_in[4];
  const float* r_q   = (const float*)d_in[5];
  const float* s_q   = (const float*)d_in[6];
  const float* r_kv  = (const float*)d_in[7];
  const float* s_kv  = (const float*)d_in[8];
  const float* r_o   = (const float*)d_in[9];
  const float* s_o   = (const float*)d_in[10];
  const float* rm_q  = (const float*)d_in[11];
  const float* sm_q  = (const float*)d_in[12];
  const float* rm_kv = (const float*)d_in[13];
  const float* sm_kv = (const float*)d_in[14];
  const float* rm_o  = (const float*)d_in[15];
  const float* sm_o  = (const float*)d_in[16];
  const int* src_idx = (const int*)d_in[17];
  const int* tgt_idx = (const int*)d_in[18];

  float* outp  = (float*)d_out;
  float* probs = outp + (long)OUT_ELEMS;   // 128 x 1024 x 1024 fp32

  // workspace layout (halfs)
  _Float16* ws  = (_Float16*)d_ws;
  _Float16* hq  = ws;                       // 8192x1024
  _Float16* hk  = hq  + 8388608;            // 8192x1024
  _Float16* wq  = hk  + 8388608;            // 1024x1024 (x SCALING)
  _Float16* wkv = wq  + 1048576;            // 2048x1024
  _Float16* wo  = wkv + 2097152;            // 1024x1024
  _Float16* qf  = wo  + 1048576;            // 8192x1024
  _Float16* kvf = qf  + 8388608;            // 8192x2048
  _Float16* vt  = kvf + 16777216;           // 128x64x1024 (v transposed)
  _Float16* ctx = vt  + 8388608;            // 8192x1024

  const float SCALING = 0.125f;  // D^-0.5, folded into Wq

  // 1) effective weights (f16)
  build_w<<<(1048576 + 255) / 256, 256, 0, stream>>>(Wq0,  r_q,  s_q,  rm_q,  sm_q,  tgt_idx, src_idx, wq,  1024, 1024, SCALING);
  build_w<<<(2097152 + 255) / 256, 256, 0, stream>>>(Wkv0, r_kv, s_kv, rm_kv, sm_kv, tgt_idx, src_idx, wkv, 2048, 1024, 1.0f);
  build_w<<<(1048576 + 255) / 256, 256, 0, stream>>>(Wo0,  r_o,  s_o,  rm_o,  sm_o,  tgt_idx, src_idx, wo,  1024, 1024, 1.0f);

  // 2) cast activations to f16
  cast_f16<<<(8388608 + 255) / 256, 256, 0, stream>>>(query, hq, 8388608L);
  cast_f16<<<(8388608 + 255) / 256, 256, 0, stream>>>(key,   hk, 8388608L);

  // 3) q = (query Wq^T) * scaling   [M=8192, N=1024, K=1024]
  gemm_bt<128, 128, _Float16><<<dim3(8, 64, 1), 256, 0, stream>>>(
      hq, 1024, wq, 1024, qf, 1024, 1024);

  // 4) kv = key Wkv^T               [M=8192, N=2048, K=1024]
  gemm_bt<128, 128, _Float16><<<dim3(16, 64, 1), 256, 0, stream>>>(
      hk, 1024, wkv, 1024, kvf, 2048, 1024);

  // 5) transpose v half of kv per head
  transpose_v<<<dim3(16, 128), 256, 0, stream>>>(kvf, vt);

  // 6) fused scores -> softmax -> ctx; probs written once to HBM
  fused_attn<<<dim3(1, 8, 128), 256, 0, stream>>>(qf, kvf, vt, probs, ctx);

  // 7) out = ctx Wo^T (fp32 out)    [M=8192, N=1024, K=1024]
  gemm_bt<128, 128, float><<<dim3(8, 64, 1), 256, 0, stream>>>(
      ctx, 1024, wo, 1024, outp, 1024, 1024);
}
